// SVDNet_51393578664643
// MI455X (gfx1250) — compile-verified, matrix operand order checked by reading, not run on hardware
//
#include <hip/hip_runtime.h>
#include <math.h>
#include <stdint.h>

// ---------------------------------------------------------------------------
// Types for CDNA5 WMMA (gfx1250, wave32)
// ---------------------------------------------------------------------------
typedef __attribute__((ext_vector_type(16))) __bf16 v16bf;
typedef __attribute__((ext_vector_type(8)))  float  v8f;

#define HW   65536   // 256*256
#define IMGW 256

// Async global->LDS path (gfx1250). Guarded so compile never fails.
#if defined(__gfx1250__) && __has_builtin(__builtin_amdgcn_global_load_async_to_lds_b32) && __has_builtin(__builtin_amdgcn_s_wait_asynccnt)
#define USE_ASYNC_LDS 1
typedef __attribute__((address_space(1))) int* g1_i32p;
typedef __attribute__((address_space(3))) int* l3_i32p;
#else
#define USE_ASYNC_LDS 0
#endif

// ---------------------------------------------------------------------------
// (M,N,2) HWC -> (2,M,N) CHW transpose
// ---------------------------------------------------------------------------
__global__ void nhwc2chw_kernel(const float* __restrict__ x, float* __restrict__ o) {
    int p = blockIdx.x * 256 + threadIdx.x;
    o[p]      = x[p * 2 + 0];
    o[HW + p] = x[p * 2 + 1];
}

// ---------------------------------------------------------------------------
// Weight prep: fp32 (Cout x K) -> bf16 (Cout x Kpad), zero padded in K.
// ---------------------------------------------------------------------------
__global__ void wprep_kernel(const float* __restrict__ w, __bf16* __restrict__ o,
                             int K, int Kpad, int total) {
    int i = blockIdx.x * 256 + threadIdx.x;
    if (i >= total) return;
    int co = i / Kpad, k = i - co * Kpad;
    o[i] = (k < K) ? (__bf16)w[(size_t)co * K + k] : (__bf16)0.0f;
}

// ---------------------------------------------------------------------------
// Implicit-GEMM convolution via v_wmma_f32_16x16x32_bf16.
//   out[co][p] = act( sum_k Wbf[co][k] * patch[k][p] + bias[co] (+ res) )
// Block: 256 threads = 8 waves. Block tile: 16 out-channels x 256 pixels.
// Each wave computes a 16x32 tile: one A fragment shared by two B fragments
// -> 2 WMMA issues per K-chunk of 32.
// A-tile (padded bf16 weights) goes through the async global->LDS mover when
// the toolchain exposes it; B-tile is an im2col gather with hoisted k-decode.
// ---------------------------------------------------------------------------
__global__ void __launch_bounds__(256)
conv_wmma_kernel(const float* __restrict__ in, const __bf16* __restrict__ wbf,
                 const float* __restrict__ bias, const float* __restrict__ res,
                 float* __restrict__ out, int Cin, int ksz, int pad, int relu) {
    __shared__ __align__(16) __bf16 At[16 * 32];     // [m][k]   1 KB
    __shared__ __align__(16) __bf16 Bt[256 * 32];    // [p][k]  16 KB

    const int tid     = threadIdx.x;
    const int coBase  = blockIdx.y * 16;
    const int pixBase = blockIdx.x * 256;
    const int khkw    = ksz * ksz;
    const int K       = Cin * khkw;
    const int Kpad    = (K + 31) & ~31;
    const int nkc     = Kpad >> 5;
    const int wave    = tid >> 5;
    const int lane    = tid & 31;
    const int kb      = (lane < 16) ? 0 : 8;   // K-half base for 16-bit operands
    const int nloc    = lane & 15;
    const int kkB     = tid & 31;              // loop-invariant K index for B fill
    const int mA      = tid >> 4;              // A-tile row for this thread
    const int qA      = tid & 15;              // A-tile dword within row

    v8f acc0 = {}, acc1 = {};

    for (int kc = 0; kc < nkc; ++kc) {
        const int k0 = kc << 5;

        // ---- A tile: 16 rows x 32 bf16 (1 KB), one b32 per thread ----
        {
            const __bf16* g = wbf + (size_t)(coBase + mA) * Kpad + k0 + qA * 2;
#if USE_ASYNC_LDS
            __builtin_amdgcn_global_load_async_to_lds_b32(
                (g1_i32p)(uintptr_t)g, (l3_i32p)&At[tid * 2], 0, 0);
#else
            ((uint32_t*)At)[tid] = *(const uint32_t*)g;
#endif
        }

        // ---- B tile: 256 pixels x 32 K, kk fixed per thread ----
        {
            const int  k      = k0 + kkB;
            const bool kvalid = (k < K);
            int ci = 0, dy = 0, dx = 0;
            if (kvalid) {
                ci = k / khkw;
                int rem = k - ci * khkw;
                int qy  = rem / ksz;
                dy = qy - pad;
                dx = (rem - qy * ksz) - pad;
            }
            const float* inc = in + (size_t)ci * HW;
            for (int p = tid >> 5; p < 256; p += 8) {
                float v = 0.0f;
                if (kvalid) {
                    int pix = pixBase + p;
                    int y = pix >> 8, x = pix & 255;
                    int yy = y + dy, xx = x + dx;
                    if (yy >= 0 && yy < IMGW && xx >= 0 && xx < IMGW)
                        v = inc[(yy << 8) + xx];
                }
                Bt[(p << 5) + kkB] = (__bf16)v;
            }
        }
#if USE_ASYNC_LDS
        __builtin_amdgcn_s_wait_asynccnt(0);
#endif
        __syncthreads();

        // ---- fragments per documented 16-bit A/B VGPR layout ----
        v16bf a, b0, b1;
        {
            const __bf16* ap = &At[((lane & 15) << 5) + kb];
#pragma unroll
            for (int i = 0; i < 8; ++i) { a[i] = ap[i]; a[8 + i] = ap[16 + i]; }
            const __bf16* bp0 = &Bt[(((wave << 5) + nloc) << 5) + kb];
#pragma unroll
            for (int i = 0; i < 8; ++i) { b0[i] = bp0[i]; b0[8 + i] = bp0[16 + i]; }
            const __bf16* bp1 = &Bt[(((wave << 5) + 16 + nloc) << 5) + kb];
#pragma unroll
            for (int i = 0; i < 8; ++i) { b1[i] = bp1[i]; b1[8 + i] = bp1[16 + i]; }
        }
        acc0 = __builtin_amdgcn_wmma_f32_16x16x32_bf16(
                   false, a, false, b0, (short)0, acc0, false, false);
        acc1 = __builtin_amdgcn_wmma_f32_16x16x32_bf16(
                   false, a, false, b1, (short)0, acc1, false, false);
        __syncthreads();
    }

    // ---- epilogue: bias (+residual) (+relu); C/D: M = v + (lane<16?0:8) ----
    const int pix0 = pixBase + (wave << 5) + nloc;
    const int pix1 = pix0 + 16;
    const int mb   = (lane < 16) ? 0 : 8;
#pragma unroll
    for (int v = 0; v < 8; ++v) {
        int   co = coBase + mb + v;
        float bv = bias[co];
        size_t oi0 = (size_t)co * HW + pix0;
        size_t oi1 = (size_t)co * HW + pix1;
        float o0 = acc0[v] + bv;
        float o1 = acc1[v] + bv;
        if (res) { o0 += res[oi0]; o1 += res[oi1]; }
        if (relu) { o0 = fmaxf(o0, 0.0f); o1 = fmaxf(o1, 0.0f); }
        out[oi0] = o0;
        out[oi1] = o1;
    }
}

// ---------------------------------------------------------------------------
// Global average pool: one block per channel
// ---------------------------------------------------------------------------
__global__ void gap_kernel(const float* __restrict__ in, float* __restrict__ out) {
    __shared__ float red[8];
    int c = blockIdx.x;
    float s = 0.0f;
    for (int p = threadIdx.x; p < HW; p += 256) s += in[(size_t)c * HW + p];
#pragma unroll
    for (int o = 16; o > 0; o >>= 1) s += __shfl_xor(s, o, 32);
    if ((threadIdx.x & 31) == 0) red[threadIdx.x >> 5] = s;
    __syncthreads();
    if (threadIdx.x == 0) {
        float t = 0.0f;
#pragma unroll
        for (int i = 0; i < 8; ++i) t += red[i];
        out[c] = t * (1.0f / (float)HW);
    }
}

// ---------------------------------------------------------------------------
// Matvec: y = act(W @ x + b), W row-major (rows x cols). One wave per row.
// ---------------------------------------------------------------------------
__global__ void matvec_kernel(const float* __restrict__ W, const float* __restrict__ x,
                              const float* __restrict__ b, float* __restrict__ y,
                              int cols, int act) {
    int wave = threadIdx.x >> 5, lane = threadIdx.x & 31;
    int row  = blockIdx.x * 8 + wave;
    const float* wr = W + (size_t)row * cols;
    float s = 0.0f;
    for (int c = lane; c < cols; c += 32) s += wr[c] * x[c];
#pragma unroll
    for (int o = 16; o > 0; o >>= 1) s += __shfl_xor(s, o, 32);
    if (lane == 0) {
        s += b[row];
        if (act == 1) s = fmaxf(s, 0.0f);
        y[row] = s;
    }
}

// ---------------------------------------------------------------------------
// softplus * |scale| then bitonic sort (descending), 64 values, 64 threads
// ---------------------------------------------------------------------------
__global__ void softsort_kernel(const float* __restrict__ sraw,
                                const float* __restrict__ scale,
                                float* __restrict__ out) {
    __shared__ float v[64];
    int t = threadIdx.x;
    float x  = sraw[t];
    float sp = (x > 20.0f) ? x : log1pf(expf(x));
    v[t] = sp * fabsf(scale[0]);
    __syncthreads();
    for (int k = 2; k <= 64; k <<= 1) {
        for (int j = k >> 1; j > 0; j >>= 1) {
            int ixj = t ^ j;
            if (ixj > t) {
                bool up = (t & k) == 0;   // descending overall
                float a = v[t], b = v[ixj];
                if (up ? (a < b) : (a > b)) { v[t] = b; v[ixj] = a; }
            }
            __syncthreads();
        }
    }
    out[t] = v[t];
}

// ---------------------------------------------------------------------------
// ctx = concat(feat768, s_values); also copy s_values into d_out slot
// ---------------------------------------------------------------------------
__global__ void ctx_kernel(const float* __restrict__ feat, const float* __restrict__ s,
                           float* __restrict__ ctx, float* __restrict__ outs) {
    int i = blockIdx.x * 256 + threadIdx.x;
    if (i < 768)      ctx[i] = feat[i];
    else if (i < 832) ctx[i] = s[i - 768];
    if (i < 64)       outs[i] = s[i];
}

// ---------------------------------------------------------------------------
// Complex modified Gram-Schmidt + 3 refinement iterations.
// One block per matrix (block 0: U, block 1: V). 256 threads = 256 rows.
// Q double-buffered in 256 KB dynamic LDS (column-major, (c*256+r)*2),
// G (64x64 complex) in 32 KB static LDS. Uses CDNA5's 320 KB LDS/WGP.
// ---------------------------------------------------------------------------
__device__ inline float2 blockReduce2(float a, float b, float* red) {
#pragma unroll
    for (int o = 16; o > 0; o >>= 1) {
        a += __shfl_xor(a, o, 32);
        b += __shfl_xor(b, o, 32);
    }
    int w = threadIdx.x >> 5;
    __syncthreads();
    if ((threadIdx.x & 31) == 0) { red[w * 2] = a; red[w * 2 + 1] = b; }
    __syncthreads();
    a = 0.0f; b = 0.0f;
#pragma unroll
    for (int i = 0; i < 8; ++i) { a += red[i * 2]; b += red[i * 2 + 1]; }
    return make_float2(a, b);
}

__global__ void __launch_bounds__(256)
gs_kernel(const float* __restrict__ Uflat, const float* __restrict__ Vflat,
          float* __restrict__ out) {
    const float* x = (blockIdx.x == 0) ? Uflat : Vflat;
    float*       o = out + ((blockIdx.x == 0) ? 0 : 32832);

    extern __shared__ float lds[];
    float* Qa = lds;            // 32768 floats (256x64 complex)
    float* Qb = lds + 32768;    // 32768 floats
    __shared__ float G[64 * 64 * 2];
    __shared__ float red[16];
    __shared__ float invn[64];

    const int t = threadIdx.x;  // row index

    // ---- modified Gram-Schmidt ----
    for (int i = 0; i < 64; ++i) {
        float qr = x[((size_t)t * 64 + i) * 2 + 0];
        float qi = x[((size_t)t * 64 + i) * 2 + 1];
        for (int j = 0; j < i; ++j) {
            float Qr = Qa[(j * 256 + t) * 2 + 0];
            float Qi = Qa[(j * 256 + t) * 2 + 1];
            float2 d = blockReduce2(Qr * qr + Qi * qi, Qr * qi - Qi * qr, red);
            qr -= d.x * Qr - d.y * Qi;
            qi -= d.x * Qi + d.y * Qr;
        }
        float2 n = blockReduce2(qr * qr + qi * qi, 0.0f, red);
        float inv = 1.0f / sqrtf(n.x + 1e-10f);
        Qa[(i * 256 + t) * 2 + 0] = qr * inv;
        Qa[(i * 256 + t) * 2 + 1] = qi * inv;
        __syncthreads();
    }

    // ---- refinement: Q -= Q @ (0.4*(Q^H Q - I)); column renorm; 3 iters ----
    float* cur = Qa;
    float* nxt = Qb;
    for (int it = 0; it < 3; ++it) {
        for (int e = t; e < 4096; e += 256) {
            int j = e >> 6, k = e & 63;
            float gr = 0.0f, gi = 0.0f;
            for (int r = 0; r < 256; ++r) {
                float ar = cur[(j * 256 + r) * 2], ai = cur[(j * 256 + r) * 2 + 1];
                float br = cur[(k * 256 + r) * 2], bi = cur[(k * 256 + r) * 2 + 1];
                gr += ar * br + ai * bi;
                gi += ar * bi - ai * br;
            }
            if (j == k) gr -= 1.0f;
            G[(j * 64 + k) * 2 + 0] = 0.4f * gr;
            G[(j * 64 + k) * 2 + 1] = 0.4f * gi;
        }
        __syncthreads();
        for (int i = 0; i < 64; ++i) {
            float ar = cur[(i * 256 + t) * 2], ai = cur[(i * 256 + t) * 2 + 1];
            for (int k = 0; k < 64; ++k) {
                float qr = cur[(k * 256 + t) * 2], qi = cur[(k * 256 + t) * 2 + 1];
                float gr = G[(k * 64 + i) * 2],    gi = G[(k * 64 + i) * 2 + 1];
                ar -= qr * gr - qi * gi;
                ai -= qr * gi + qi * gr;
            }
            nxt[(i * 256 + t) * 2 + 0] = ar;
            nxt[(i * 256 + t) * 2 + 1] = ai;
        }
        __syncthreads();
        for (int c = t; c < 64; c += 256) {
            float s = 0.0f;
            for (int r = 0; r < 256; ++r) {
                float ar = nxt[(c * 256 + r) * 2], ai = nxt[(c * 256 + r) * 2 + 1];
                s += ar * ar + ai * ai;
            }
            invn[c] = 1.0f / sqrtf(s + 1e-10f);
        }
        __syncthreads();
        for (int e = t; e < 16384; e += 256) {
            int c = e >> 8;
            nxt[e * 2 + 0] *= invn[c];
            nxt[e * 2 + 1] *= invn[c];
        }
        __syncthreads();
        float* tmp = cur; cur = nxt; nxt = tmp;
    }

    // ---- write out as (row, col, 2) ----
    for (int e = t; e < 16384; e += 256) {
        int r = e & 255, c = e >> 8;
        o[(r * 64 + c) * 2 + 0] = cur[e * 2 + 0];
        o[(r * 64 + c) * 2 + 1] = cur[e * 2 + 1];
    }
}

// ---------------------------------------------------------------------------
// Orchestration
// ---------------------------------------------------------------------------
extern "C" void kernel_launch(void* const* d_in, const int* in_sizes, int n_in,
                              void* d_out, int out_size, void* d_ws, size_t ws_size,
                              hipStream_t stream) {
    const float* x      = (const float*)d_in[0];
    const float* c0_w   = (const float*)d_in[1];   const float* c0_b   = (const float*)d_in[2];
    const float* r1c1_w = (const float*)d_in[3];   const float* r1c1_b = (const float*)d_in[4];
    const float* r1c2_w = (const float*)d_in[5];   const float* r1c2_b = (const float*)d_in[6];
    const float* r1s_w  = (const float*)d_in[7];   const float* r1s_b  = (const float*)d_in[8];
    const float* r2c1_w = (const float*)d_in[9];   const float* r2c1_b = (const float*)d_in[10];
    const float* r2c2_w = (const float*)d_in[11];  const float* r2c2_b = (const float*)d_in[12];
    const float* r2s_w  = (const float*)d_in[13];  const float* r2s_b  = (const float*)d_in[14];
    const float* c3_w   = (const float*)d_in[15];  const float* c3_b   = (const float*)d_in[16];
    const float* f1_w   = (const float*)d_in[17];  const float* f1_b   = (const float*)d_in[18];
    const float* f2_w   = (const float*)d_in[19];  const float* f2_b   = (const float*)d_in[20];
    const float* s1_w   = (const float*)d_in[21];  const float* s1_b   = (const float*)d_in[22];
    const float* s2_w   = (const float*)d_in[23];  const float* s2_b   = (const float*)d_in[24];
    const float* s3_w   = (const float*)d_in[25];  const float* s3_b   = (const float*)d_in[26];
    const float* u1_w   = (const float*)d_in[27];  const float* u1_b   = (const float*)d_in[28];
    const float* u2_w   = (const float*)d_in[29];  const float* u2_b   = (const float*)d_in[30];
    const float* v1_w   = (const float*)d_in[31];  const float* v1_b   = (const float*)d_in[32];
    const float* v2_w   = (const float*)d_in[33];  const float* v2_b   = (const float*)d_in[34];
    const float* s_sc   = (const float*)d_in[35];

    float* ws = (float*)d_ws;
    float* P0 = ws;                    // 256ch pool
    float* P1 = ws + 16777216;         // 256ch pool
    float* P2 = ws + 33554432;         // 256ch pool
    float* P3 = ws + 50331648;         // 512ch pool
    float* SM = ws + 83886080;         // small buffers
    float* feat = SM + 0;     float* f1o = SM + 512;   float* f2o = SM + 1536;
    float* s1o  = SM + 2304;  float* s2o = SM + 2816;  float* s3r = SM + 3072;
    float* ssv  = SM + 3136;  float* ctx = SM + 3200;
    float* hu   = SM + 4032;  float* hv  = SM + 5056;
    float* Uf   = SM + 6080;  float* Vf  = SM + 38848;
    // bf16 padded weights region
    __bf16* WB      = (__bf16*)(ws + 83966080);
    __bf16* wb_c0   = WB + 0;        // 64 x 32     = 2048
    __bf16* wb_r1s  = WB + 2048;     // 128 x 64    = 8192
    __bf16* wb_r1c1 = WB + 10240;    // 128 x 576   = 73728
    __bf16* wb_r1c2 = WB + 83968;    // 128 x 1152  = 147456
    __bf16* wb_r2s  = WB + 231424;   // 256 x 128   = 32768
    __bf16* wb_r2c1 = WB + 264192;   // 256 x 1152  = 294912
    __bf16* wb_r2c2 = WB + 559104;   // 256 x 2304  = 589824
    __bf16* wb_c3   = WB + 1148928;  // 512 x 2304  = 1179648
    float* outp = (float*)d_out;

    dim3 blk(256);
    nhwc2chw_kernel<<<256, blk, 0, stream>>>(x, P0);

    // ---- weight prep to padded bf16 ----
    wprep_kernel<<<8,    blk, 0, stream>>>(c0_w,   wb_c0,   18,   32,   2048);
    wprep_kernel<<<32,   blk, 0, stream>>>(r1s_w,  wb_r1s,  64,   64,   8192);
    wprep_kernel<<<288,  blk, 0, stream>>>(r1c1_w, wb_r1c1, 576,  576,  73728);
    wprep_kernel<<<576,  blk, 0, stream>>>(r1c2_w, wb_r1c2, 1152, 1152, 147456);
    wprep_kernel<<<128,  blk, 0, stream>>>(r2s_w,  wb_r2s,  128,  128,  32768);
    wprep_kernel<<<1152, blk, 0, stream>>>(r2c1_w, wb_r2c1, 1152, 1152, 294912);
    wprep_kernel<<<2304, blk, 0, stream>>>(r2c2_w, wb_r2c2, 2304, 2304, 589824);
    wprep_kernel<<<4608, blk, 0, stream>>>(c3_w,   wb_c3,   2304, 2304, 1179648);

    // ---- conv stack (grid.x = 65536/256 pixel tiles, grid.y = Cout/16) ----
    conv_wmma_kernel<<<dim3(256, 4),  blk, 0, stream>>>(P0, wb_c0,   c0_b,   nullptr, P1, 2,   3, 1, 1);
    conv_wmma_kernel<<<dim3(256, 8),  blk, 0, stream>>>(P1, wb_r1s,  r1s_b,  nullptr, P2, 64,  1, 0, 0);
    conv_wmma_kernel<<<dim3(256, 8),  blk, 0, stream>>>(P1, wb_r1c1, r1c1_b, nullptr, P0, 64,  3, 1, 1);
    conv_wmma_kernel<<<dim3(256, 8),  blk, 0, stream>>>(P0, wb_r1c2, r1c2_b, P2,      P1, 128, 3, 1, 1);
    conv_wmma_kernel<<<dim3(256, 16), blk, 0, stream>>>(P1, wb_r2s,  r2s_b,  nullptr, P2, 128, 1, 0, 0);
    conv_wmma_kernel<<<dim3(256, 16), blk, 0, stream>>>(P1, wb_r2c1, r2c1_b, nullptr, P0, 128, 3, 1, 1);
    conv_wmma_kernel<<<dim3(256, 16), blk, 0, stream>>>(P0, wb_r2c2, r2c2_b, P2,      P1, 256, 3, 1, 1);
    conv_wmma_kernel<<<dim3(256, 32), blk, 0, stream>>>(P1, wb_c3,   c3_b,   nullptr, P3, 256, 3, 1, 1);

    gap_kernel<<<512, blk, 0, stream>>>(P3, feat);

    matvec_kernel<<<128, blk, 0, stream>>>(f1_w, feat, f1_b, f1o, 512, 1);
    matvec_kernel<<<96,  blk, 0, stream>>>(f2_w, f1o,  f2_b, f2o, 1024, 1);
    matvec_kernel<<<64,  blk, 0, stream>>>(s1_w, f2o,  s1_b, s1o, 768, 1);
    matvec_kernel<<<32,  blk, 0, stream>>>(s2_w, s1o,  s2_b, s2o, 512, 1);
    matvec_kernel<<<8,   blk, 0, stream>>>(s3_w, s2o,  s3_b, s3r, 256, 0);

    softsort_kernel<<<1, 64, 0, stream>>>(s3r, s_sc, ssv);
    ctx_kernel<<<4, blk, 0, stream>>>(f2o, ssv, ctx, outp + 32768);

    matvec_kernel<<<128,  blk, 0, stream>>>(u1_w, ctx, u1_b, hu, 832, 1);
    matvec_kernel<<<4096, blk, 0, stream>>>(u2_w, hu,  u2_b, Uf, 1024, 0);
    matvec_kernel<<<128,  blk, 0, stream>>>(v1_w, ctx, v1_b, hv, 832, 1);
    matvec_kernel<<<4096, blk, 0, stream>>>(v2_w, hv,  v2_b, Vf, 1024, 0);

    gs_kernel<<<2, blk, 262144, stream>>>(Uf, Vf, outp);
}
